// MMDLoss_11304353923116
// MI455X (gfx1250) — matrix operand order repaired and checked by reading, not compiled, not run on hardware
//
#include <hip/hip_runtime.h>

// ---------------------------------------------------------------------------
// MMD loss, fused:  G = X·Xᵀ via split-bf16 WMMA (hi·hi + hi·lo + lo·hi),
// bandwidth computed analytically, exp-kernel + signed reduction fused into
// the GEMM epilogue. Deterministic (no atomics, fixed-order reductions).
// ---------------------------------------------------------------------------

typedef __attribute__((ext_vector_type(16))) __bf16 v16bf;
typedef __attribute__((ext_vector_type(8)))  __bf16 v8bf;
typedef __attribute__((ext_vector_type(8)))  float  v8f;

#define N_ROWS 4096
#define DIM    1024
#define LTOT   8192
#define LDSTR  40   // padded bf16 row stride for a 32-element K chunk (80 B)

// ---- WMMA wrapper ---------------------------------------------------------
__device__ __forceinline__ v8f wmma_bf16(v16bf a, v16bf b, v8f c) {
  return __builtin_amdgcn_wmma_f32_16x16x32_bf16(
      /*neg_a=*/false, a, /*neg_b=*/false, b,
      /*c_mod=*/(short)0, c, /*reuse_a=*/false, /*reuse_b=*/false);
}

// ---- Fragment gathers from LDS (layouts per CDNA5 ISA 7.12.2) -------------
// A 16x32 bf16: lanes 0-15 row m hold K 0-7 (v0-3) and 16-23 (v4-7);
//               lanes 16-31 hold K 8-15 and 24-31.
__device__ __forceinline__ v16bf frag_a(const __bf16* base, int rowTile, int lane) {
  const int m  = rowTile + (lane & 15);
  const int kb = (lane < 16) ? 0 : 8;
  const __bf16* p = base + m * LDSTR + kb;
  v8bf x0 = *(const v8bf*)(p);        // K = kb .. kb+7
  v8bf x1 = *(const v8bf*)(p + 16);   // K = kb+16 .. kb+23
  return __builtin_shufflevector(x0, x1, 0,1,2,3,4,5,6,7,8,9,10,11,12,13,14,15);
}
// B 32x16 bf16: lanes 0-15 col n hold K 0-7 and 16-23; lanes 16-31 hold K 8-15 and 24-31.
__device__ __forceinline__ v16bf frag_b(const __bf16* base, int colTile, int lane) {
  const int nn = colTile + (lane & 15);
  const int kb = (lane < 16) ? 0 : 8;
  const __bf16* p = base + nn * LDSTR + kb;
  v8bf x0 = *(const v8bf*)(p);        // K = kb .. kb+7
  v8bf x1 = *(const v8bf*)(p + 16);   // K = kb+16 .. kb+23
  return __builtin_shufflevector(x0, x1, 0,1,2,3,4,5,6,7,8,9,10,11,12,13,14,15);
}

// ---- fp32 -> (bf16 hi, bf16 lo) split, 16 elems, wide LDS stores ----------
__device__ __forceinline__ void cvt_store16(const float* g, __bf16* dhi, __bf16* dlo) {
  const float4* gp = (const float4*)g;
  float4 q0 = gp[0], q1 = gp[1], q2 = gp[2], q3 = gp[3];
  float xs[16] = {q0.x,q0.y,q0.z,q0.w, q1.x,q1.y,q1.z,q1.w,
                  q2.x,q2.y,q2.z,q2.w, q3.x,q3.y,q3.z,q3.w};
  v8bf h0, h1, l0, l1;
#pragma unroll
  for (int i = 0; i < 8; ++i) {
    __bf16 a = (__bf16)xs[i];
    h0[i] = a;  l0[i] = (__bf16)(xs[i] - (float)a);
    __bf16 b = (__bf16)xs[i + 8];
    h1[i] = b;  l1[i] = (__bf16)(xs[i + 8] - (float)b);
  }
  *(v8bf*)(dhi)     = h0;  *(v8bf*)(dhi + 8) = h1;
  *(v8bf*)(dlo)     = l0;  *(v8bf*)(dlo + 8) = l1;
}

// ---------------------------------------------------------------------------
// Kernel 1: per-row sum of squares  sq[i] = ||x_i||^2  (fp32, like reference)
// ---------------------------------------------------------------------------
__global__ void row_sq_kernel(const float* __restrict__ src,
                              const float* __restrict__ tgt,
                              float* __restrict__ sq) {
  const int row = blockIdx.x;
  const float* p = (row < N_ROWS) ? src + (size_t)row * DIM
                                  : tgt + (size_t)(row - N_ROWS) * DIM;
  float s = 0.f;
  for (int k = threadIdx.x; k < DIM; k += 256) { float x = p[k]; s += x * x; }
  __shared__ float red[256];
  red[threadIdx.x] = s; __syncthreads();
  for (int off = 128; off > 0; off >>= 1) {
    if (threadIdx.x < off) red[threadIdx.x] += red[threadIdx.x + off];
    __syncthreads();
  }
  if (threadIdx.x == 0) sq[row] = red[0];
}

// ---------------------------------------------------------------------------
// Kernel 2: partial column sums (for ||sum_i x_i||^2).  32 blocks: block b
// covers k-range (b%4)*256..+255 over row-chunk (b/4)*1024..+1023.
// ---------------------------------------------------------------------------
__global__ void col_part_kernel(const float* __restrict__ src,
                                const float* __restrict__ tgt,
                                double* __restrict__ colpart) {
  const int k     = (blockIdx.x & 3) * 256 + threadIdx.x;   // 0..1023
  const int chunk = blockIdx.x >> 2;                        // 0..7
  const int r0    = chunk * 1024;                           // global row base
  double s0 = 0.0, s1 = 0.0, s2 = 0.0, s3 = 0.0;
  for (int i = 0; i < 1024; i += 4) {
    const int row = r0 + i;
    const float* p = (row < N_ROWS) ? src + (size_t)row * DIM
                                    : tgt + (size_t)(row - N_ROWS) * DIM;
    s0 += (double)p[k];
    s1 += (double)p[k + DIM];
    s2 += (double)p[k + 2 * DIM];
    s3 += (double)p[k + 3 * DIM];
  }
  colpart[(size_t)chunk * DIM + k] = (s0 + s1) + (s2 + s3);
}

// ---------------------------------------------------------------------------
// Kernel 3: bandwidth.  sum_d2 = 2*L*S1 - 2*||v||^2  (diagonal is exactly 0,
// off-diagonal d2 >> 0 so the clamp is a no-op analytically).
// bw = sum_d2/(L^2-L) / KERNEL_MUL^(KERNEL_NUM/2) = .../4
// ---------------------------------------------------------------------------
__global__ void bw_kernel(const float* __restrict__ sq,
                          const double* __restrict__ colpart,
                          double* __restrict__ bwp) {
  const int tid = threadIdx.x;
  double s1 = 0.0, vsq = 0.0;
  for (int i = tid; i < LTOT; i += 256) s1 += (double)sq[i];
  for (int k = tid; k < DIM; k += 256) {
    double c = 0.0;
#pragma unroll
    for (int ch = 0; ch < 8; ++ch) c += colpart[(size_t)ch * DIM + k];
    vsq += c * c;
  }
  __shared__ double r1[256], r2[256];
  r1[tid] = s1; r2[tid] = vsq; __syncthreads();
  for (int off = 128; off > 0; off >>= 1) {
    if (tid < off) { r1[tid] += r1[tid + off]; r2[tid] += r2[tid + off]; }
    __syncthreads();
  }
  if (tid == 0) {
    const double L = (double)LTOT;
    double sum_d2 = 2.0 * L * r1[0] - 2.0 * r2[0];
    bwp[0] = sum_d2 / (L * L - L) / 4.0;
  }
}

// ---------------------------------------------------------------------------
// Kernel 4: fused GEMM (split-bf16 WMMA) + RBF kernel + signed block reduce.
// 256 threads = 8 wave32; block tile 128x128; wave tile 64x32 (4x2 WMMAs).
// ---------------------------------------------------------------------------
__global__ void __launch_bounds__(256)
mmd_main_kernel(const float* __restrict__ src, const float* __restrict__ tgt,
                const float* __restrict__ sq, const double* __restrict__ bwp,
                double* __restrict__ partials) {
  const int tid  = threadIdx.x;
  const int lane = tid & 31;
  const int wave = tid >> 5;
  const int wM   = (wave >> 2) * 64;   // 2 waves along M
  const int wN   = (wave & 3) * 32;    // 4 waves along N
  const int rowBase = blockIdx.y * 128;
  const int colBase = blockIdx.x * 128;
  // 128 | 4096, so a block lies entirely in one half for rows and for cols.
  const float* rowPtr = (rowBase < N_ROWS) ? src + (size_t)rowBase * DIM
                                           : tgt + (size_t)(rowBase - N_ROWS) * DIM;
  const float* colPtr = (colBase < N_ROWS) ? src + (size_t)colBase * DIM
                                           : tgt + (size_t)(colBase - N_ROWS) * DIM;

  __shared__ __align__(16) __bf16 lAhi[128 * LDSTR];
  __shared__ __align__(16) __bf16 lAlo[128 * LDSTR];
  __shared__ __align__(16) __bf16 lBhi[128 * LDSTR];
  __shared__ __align__(16) __bf16 lBlo[128 * LDSTR];
  __shared__ float  sqRow[128], sqCol[128];
  __shared__ double red[256];

  if (tid < 128) { sqRow[tid] = sq[rowBase + tid]; sqCol[tid] = sq[colBase + tid]; }
  const float invbw = (float)(1.0 / bwp[0]);

  v8f acc[4][2];
  const v8f zero = {};
#pragma unroll
  for (int m = 0; m < 4; ++m)
#pragma unroll
    for (int nq = 0; nq < 2; ++nq) acc[m][nq] = zero;

  const int r = tid >> 1;   // local row this thread stages (0..127)
  const int h = tid & 1;    // which 16-element half of the 32-wide K chunk

  for (int kc = 0; kc < DIM; kc += 32) {
    __syncthreads();  // previous iteration's fragment reads done
    cvt_store16(rowPtr + (size_t)r * DIM + kc + h * 16,
                &lAhi[r * LDSTR + h * 16], &lAlo[r * LDSTR + h * 16]);
    cvt_store16(colPtr + (size_t)r * DIM + kc + h * 16,
                &lBhi[r * LDSTR + h * 16], &lBlo[r * LDSTR + h * 16]);
    __syncthreads();

    v16bf bhi0 = frag_b(lBhi, wN,      lane);
    v16bf bhi1 = frag_b(lBhi, wN + 16, lane);
    v16bf blo0 = frag_b(lBlo, wN,      lane);
    v16bf blo1 = frag_b(lBlo, wN + 16, lane);
#pragma unroll
    for (int mt = 0; mt < 4; ++mt) {
      v16bf ahi = frag_a(lAhi, wM + mt * 16, lane);
      v16bf alo = frag_a(lAlo, wM + mt * 16, lane);
      acc[mt][0] = wmma_bf16(ahi, bhi0, acc[mt][0]);  // hi*hi
      acc[mt][0] = wmma_bf16(ahi, blo0, acc[mt][0]);  // hi*lo
      acc[mt][0] = wmma_bf16(alo, bhi0, acc[mt][0]);  // lo*hi
      acc[mt][1] = wmma_bf16(ahi, bhi1, acc[mt][1]);
      acc[mt][1] = wmma_bf16(ahi, blo1, acc[mt][1]);
      acc[mt][1] = wmma_bf16(alo, bhi1, acc[mt][1]);
    }
  }

  // Epilogue: d2 -> 5-bandwidth RBF -> per-thread double accumulation.
  // C layout: VGPR v -> (M = v + (lane<16?0:8), N = lane&15).
  double accd = 0.0;
#pragma unroll
  for (int mt = 0; mt < 4; ++mt) {
    const int ib = wM + mt * 16 + ((lane < 16) ? 0 : 8);
#pragma unroll
    for (int nt = 0; nt < 2; ++nt) {
      const float sqj = sqCol[wN + nt * 16 + (lane & 15)];
#pragma unroll
      for (int v = 0; v < 8; ++v) {
        const float sqi = sqRow[ib + v];
        const float g   = acc[mt][nt][v];
        const float d2  = fmaxf(sqi + sqj - 2.0f * g, 0.0f);
        const float t   = d2 * invbw;
        const float K   = __expf(-t)            + __expf(-t * 0.5f)
                        + __expf(-t * 0.25f)    + __expf(-t * 0.125f)
                        + __expf(-t * 0.0625f);
        accd += (double)K;
      }
    }
  }

  red[tid] = accd; __syncthreads();
  for (int off = 128; off > 0; off >>= 1) {
    if (tid < off) red[tid] += red[tid + off];
    __syncthreads();
  }
  if (tid == 0) {
    const double sgn = ((rowBase < N_ROWS) == (colBase < N_ROWS)) ? 1.0 : -1.0;
    partials[(size_t)blockIdx.y * 64 + blockIdx.x] = sgn * red[0];
  }
}

// ---------------------------------------------------------------------------
// Kernel 5: final deterministic reduction of 4096 block partials.
// ---------------------------------------------------------------------------
__global__ void final_kernel(const double* __restrict__ partials,
                             float* __restrict__ out) {
  const int tid = threadIdx.x;
  double s = 0.0;
  for (int i = tid; i < 4096; i += 256) s += partials[i];
  __shared__ double red[256];
  red[tid] = s; __syncthreads();
  for (int off = 128; off > 0; off >>= 1) {
    if (tid < off) red[tid] += red[tid + off];
    __syncthreads();
  }
  if (tid == 0)
    out[0] = (float)(red[0] / (5.0 * (double)N_ROWS * (double)N_ROWS));
}

// ---------------------------------------------------------------------------
extern "C" void kernel_launch(void* const* d_in, const int* in_sizes, int n_in,
                              void* d_out, int out_size, void* d_ws, size_t ws_size,
                              hipStream_t stream) {
  const float* src = (const float*)d_in[0];
  const float* tgt = (const float*)d_in[1];
  float* out = (float*)d_out;

  // Workspace layout (131080 bytes total):
  char* ws = (char*)d_ws;
  float*  sqf      = (float*) (ws);              // 8192 f32   -> 32768 B
  double* colpart  = (double*)(ws + 32768);      // 8*1024 f64 -> 65536 B
  double* bwp      = (double*)(ws + 98304);      // 1 f64      ->     8 B
  double* partials = (double*)(ws + 98312);      // 4096 f64   -> 32768 B

  row_sq_kernel  <<<LTOT, 256, 0, stream>>>(src, tgt, sqf);
  col_part_kernel<<<32,   256, 0, stream>>>(src, tgt, colpart);
  bw_kernel      <<<1,    256, 0, stream>>>(sqf, colpart, bwp);
  mmd_main_kernel<<<dim3(64, 64), 256, 0, stream>>>(src, tgt, sqf, bwp, partials);
  final_kernel   <<<1,    256, 0, stream>>>(partials, out);
}
